// InfiniteContextMemory_12360915878584
// MI455X (gfx1250) — matrix operand order repaired and checked by reading, not compiled, not run on hardware
//
#include <hip/hip_runtime.h>
#include <math.h>

// ---------------------------------------------------------------------------
// CDNA5 (gfx1250) implementation: bf16 WMMA GEMMs + flash attention.
// Wave32. All matmuls use v_wmma_f32_16x16x32_bf16 (fp32 accumulate).
// ---------------------------------------------------------------------------

typedef __attribute__((ext_vector_type(16))) __bf16 v16bf;
typedef __attribute__((ext_vector_type(2)))  __bf16 v2bf;
typedef __attribute__((ext_vector_type(8)))  float  v8f;
typedef __attribute__((ext_vector_type(2)))  float  v2f;

union BF16Frag { uint4 q[2]; v16bf v; };

// native bf16 convert (RNE) -- lowers to v_cvt_*bf16_f32 on gfx1250
__device__ __forceinline__ unsigned short f2bf(float f) {
  union { __bf16 h; unsigned short u; } c;
  c.h = (__bf16)f;
  return c.u;
}

// packed 4-float -> 4-bf16 convert + single 8B store (v_cvt_pk_bf16_f32 x2)
__device__ __forceinline__ void store4bf(unsigned short* p, float4 f) {
  v2f a = {f.x, f.y};
  v2f b = {f.z, f.w};
  union { v2bf v; unsigned int u; } ca, cb;
  ca.v = __builtin_convertvector(a, v2bf);
  cb.v = __builtin_convertvector(b, v2bf);
  uint2 packed; packed.x = ca.u; packed.y = cb.u;
  *reinterpret_cast<uint2*>(p) = packed;   // 8B aligned (row stride 80B, col4*8B)
}

// ---------------------------------------------------------------------------
// GEMM: C[M,N] = act(A[M,K] @ W[N,K]^T + bias[N])
// Block: 256 threads = 8 waves, tile 128(M) x 64(N), K step 32.
// Wave (wm in 0..3, wn in 0..1) computes 32x32 via 2x2 WMMA fragments.
// LDS row stride 40 bf16 (80B): 16B-aligned b128 loads, no bank conflicts.
// ---------------------------------------------------------------------------
#define LDA 40

__device__ __forceinline__ v16bf lds_a_frag(const unsigned short* base, int mOff, int lane) {
  int m  = mOff + (lane & 15);
  int kb = (lane < 16) ? 0 : 8;          // A layout: low lanes K=0.., high lanes K=8..
  const unsigned short* p = base + m * LDA + kb;
  BF16Frag f;
  f.q[0] = *reinterpret_cast<const uint4*>(p);        // K = kb .. kb+7
  f.q[1] = *reinterpret_cast<const uint4*>(p + 16);   // K = kb+16 .. kb+23
  return f.v;
}

__device__ __forceinline__ v16bf lds_b_frag(const unsigned short* base, int nOff, int lane) {
  int n  = nOff + (lane & 15);
  int kb = (lane < 16) ? 0 : 8;          // B layout: low lanes K=0..7,16..23, high lanes K=8..15,24..31
  const unsigned short* p = base + n * LDA + kb;
  BF16Frag f;
  f.q[0] = *reinterpret_cast<const uint4*>(p);        // K = kb .. kb+7
  f.q[1] = *reinterpret_cast<const uint4*>(p + 16);   // K = kb+16 .. kb+23
  return f.v;
}

__global__ __launch_bounds__(256)
void gemm_bias_kernel(const float* __restrict__ A, const float* __restrict__ W,
                      const float* __restrict__ bias, float* __restrict__ C,
                      int M, int N, int K, int relu) {
  __shared__ __align__(16) unsigned short sA[128 * LDA];
  __shared__ __align__(16) unsigned short sB[64 * LDA];

  const int tid  = threadIdx.x;
  const int lane = tid & 31;
  const int wave = tid >> 5;
  const int wm   = wave >> 1;        // 0..3
  const int wn   = wave & 1;         // 0..1
  const int mBlk = blockIdx.y * 128;
  const int nBlk = blockIdx.x * 64;

  const v8f vzero = {0.f,0.f,0.f,0.f,0.f,0.f,0.f,0.f};
  v8f c00 = vzero, c01 = vzero, c10 = vzero, c11 = vzero;

  const int col4 = tid & 7;          // float4 column (8 per 32-wide k tile)
  const int row0 = tid >> 3;         // 0..31

  // W rows: clamp OOB rows to N-1 (valid address) and zero via mask -> all
  // loads unconditional, clause together with A loads.
  float wMask[2];
  const float* srcW0[2];
#pragma unroll
  for (int rr = 0; rr < 2; ++rr) {
    int n = nBlk + row0 + rr * 32;
    wMask[rr] = (n < N) ? 1.f : 0.f;
    int nc = (n < N) ? n : (N - 1);
    srcW0[rr] = W + (size_t)nc * K + col4 * 4;
  }

  for (int k0 = 0; k0 < K; k0 += 32) {
    // -- pass 1: issue ALL global loads first (one clause, full MLP) --------
    const float* srcA[4];
    float4 fa[4], fw[2];
#pragma unroll
    for (int rr = 0; rr < 4; ++rr) {
      int r = row0 + rr * 32;
      srcA[rr] = A + (size_t)(mBlk + r) * K + k0 + col4 * 4;
      fa[rr] = *reinterpret_cast<const float4*>(srcA[rr]);
    }
#pragma unroll
    for (int rr = 0; rr < 2; ++rr)
      fw[rr] = *reinterpret_cast<const float4*>(srcW0[rr] + k0);
    // -- pass 2: single-branch prefetch of next K tile ----------------------
    if (k0 + 32 < K) {
#pragma unroll
      for (int rr = 0; rr < 4; ++rr) __builtin_prefetch(srcA[rr] + 32, 0, 1);
#pragma unroll
      for (int rr = 0; rr < 2; ++rr) __builtin_prefetch(srcW0[rr] + k0 + 32, 0, 1);
    }
    // -- pass 3: mask + packed convert + LDS store --------------------------
#pragma unroll
    for (int rr = 0; rr < 4; ++rr)
      store4bf(&sA[(row0 + rr * 32) * LDA + col4 * 4], fa[rr]);
#pragma unroll
    for (int rr = 0; rr < 2; ++rr) {
      float4 f = fw[rr];
      f.x *= wMask[rr]; f.y *= wMask[rr]; f.z *= wMask[rr]; f.w *= wMask[rr];
      store4bf(&sB[(row0 + rr * 32) * LDA + col4 * 4], f);
    }
    __syncthreads();

    v16bf a0 = lds_a_frag(sA, wm * 32,      lane);
    v16bf a1 = lds_a_frag(sA, wm * 32 + 16, lane);
    v16bf b0 = lds_b_frag(sB, wn * 32,      lane);
    v16bf b1 = lds_b_frag(sB, wn * 32 + 16, lane);

    c00 = __builtin_amdgcn_wmma_f32_16x16x32_bf16(false, a0, false, b0, (short)0, c00, false, false);
    c01 = __builtin_amdgcn_wmma_f32_16x16x32_bf16(false, a0, false, b1, (short)0, c01, false, false);
    c10 = __builtin_amdgcn_wmma_f32_16x16x32_bf16(false, a1, false, b0, (short)0, c10, false, false);
    c11 = __builtin_amdgcn_wmma_f32_16x16x32_bf16(false, a1, false, b1, (short)0, c11, false, false);
    __syncthreads();
  }

  // store: C layout VGPR r -> rows r (lanes 0-15) / r+8 (lanes 16-31), col = lane&15
  const int hi = lane >> 4;
  const int cl = lane & 15;
#pragma unroll
  for (int fm = 0; fm < 2; ++fm) {
#pragma unroll
    for (int fn = 0; fn < 2; ++fn) {
      v8f cf = (fm == 0) ? ((fn == 0) ? c00 : c01) : ((fn == 0) ? c10 : c11);
      int ncol = nBlk + wn * 32 + fn * 16 + cl;
      if (ncol < N) {
        float bv = bias[ncol];
        int mrowBase = mBlk + wm * 32 + fm * 16 + hi * 8;
#pragma unroll
        for (int r = 0; r < 8; ++r) {
          float v = cf[r] + bv;
          if (relu) v = fmaxf(v, 0.f);
          C[(size_t)(mrowBase + r) * N + ncol] = v;
        }
      }
    }
  }
}

// ---------------------------------------------------------------------------
// Flash attention. Grid: (S/64, dvSlices, B*H). Block: 128 (4 waves).
// Wave handles 16 q-rows, streams 32-key tiles, dv slice = NC*16 (<=128).
// Q/K bf16 [z][S][hdq]; Vt bf16 [z][dvTotal][S]; O f32 row stride oLd.
// ---------------------------------------------------------------------------
template <int NC, bool MASK>
__global__ __launch_bounds__(128)
void flash_kernel(const unsigned short* __restrict__ Q,
                  const unsigned short* __restrict__ Kb,
                  const unsigned short* __restrict__ Vt,
                  float* __restrict__ O,
                  int S, int hdq, int dvTotal, int H, int oLd,
                  float scale, int wHalf, int strideJ) {
  const int lane = threadIdx.x & 31;
  const int wave = threadIdx.x >> 5;
  const int z = blockIdx.z;
  const int b = z / H, h = z % H;
  const int q0 = blockIdx.x * 64 + wave * 16;
  const int dvOff = blockIdx.y * (NC * 16);

  const unsigned short* Qz = Q  + (size_t)z * S * hdq;
  const unsigned short* Kz = Kb + (size_t)z * S * hdq;
  const unsigned short* Vz = Vt + (size_t)z * dvTotal * S;

  __shared__ __align__(16) unsigned short pbuf[4 * 16 * LDA];
  unsigned short* pw = pbuf + wave * 16 * LDA;

  const int hi = lane >> 4;
  const int cl = lane & 15;
  const v8f vzero = {0.f,0.f,0.f,0.f,0.f,0.f,0.f,0.f};

  v8f acc[NC];
#pragma unroll
  for (int c = 0; c < NC; ++c) acc[c] = vzero;
  float mrun[8], lrun[8];
#pragma unroll
  for (int r = 0; r < 8; ++r) { mrun[r] = -1e30f; lrun[r] = 0.f; }

  for (int kt = 0; kt < S; kt += 32) {
    v8f s0 = vzero, s1 = vzero;
    for (int kc = 0; kc < hdq; kc += 32) {
      BF16Frag fa, fb0, fb1;
      { // Q A-fragment (direct global, bf16 row-major)
        int kb = hi ? 8 : 0;
        const unsigned short* p = Qz + (size_t)(q0 + cl) * hdq + kc + kb;
        fa.q[0] = *reinterpret_cast<const uint4*>(p);
        fa.q[1] = *reinterpret_cast<const uint4*>(p + 16);
      }
      { // K B-fragments: lane carries key (N), VGPRs carry hd (K)
        int kb = hi ? 8 : 0;
        const unsigned short* p0 = Kz + (size_t)(kt + cl) * hdq + kc + kb;
        fb0.q[0] = *reinterpret_cast<const uint4*>(p0);
        fb0.q[1] = *reinterpret_cast<const uint4*>(p0 + 16);
        const unsigned short* p1 = Kz + (size_t)(kt + 16 + cl) * hdq + kc + kb;
        fb1.q[0] = *reinterpret_cast<const uint4*>(p1);
        fb1.q[1] = *reinterpret_cast<const uint4*>(p1 + 16);
      }
      s0 = __builtin_amdgcn_wmma_f32_16x16x32_bf16(false, fa.v, false, fb0.v, (short)0, s0, false, false);
      s1 = __builtin_amdgcn_wmma_f32_16x16x32_bf16(false, fa.v, false, fb1.v, (short)0, s1, false, false);
    }

    // online softmax (rows live in 16-lane halves; shfl_xor <16 stays in half)
#pragma unroll
    for (int r = 0; r < 8; ++r) {
      int i = q0 + r + hi * 8;
      float v0 = s0[r] * scale;
      float v1 = s1[r] * scale;
      if (MASK) {
        int j0 = kt + cl, j1 = kt + 16 + cl;
        bool ok0 = ((j0 >= i - wHalf) && (j0 < i + wHalf)) || ((j0 % strideJ) == 0);
        bool ok1 = ((j1 >= i - wHalf) && (j1 < i + wHalf)) || ((j1 % strideJ) == 0);
        if (!ok0) v0 = -1e30f;
        if (!ok1) v1 = -1e30f;
      }
      float mx = fmaxf(v0, v1);
#pragma unroll
      for (int msk = 1; msk < 16; msk <<= 1) mx = fmaxf(mx, __shfl_xor(mx, msk, 32));
      mx = fmaxf(mx, mrun[r]);
      float corr = __expf(mrun[r] - mx);
      float p0 = __expf(v0 - mx);
      float p1 = __expf(v1 - mx);
      float rs = p0 + p1;
#pragma unroll
      for (int msk = 1; msk < 16; msk <<= 1) rs += __shfl_xor(rs, msk, 32);
      lrun[r] = lrun[r] * corr + rs;
      mrun[r] = mx;
#pragma unroll
      for (int c = 0; c < NC; ++c) acc[c][r] *= corr;
      int row2 = r + hi * 8;
      pw[row2 * LDA + cl]      = f2bf(p0);
      pw[row2 * LDA + 16 + cl] = f2bf(p1);
    }

    // reload P as A-fragment (LDS ops in-order within wave)
    BF16Frag fp;
    {
      int kb = hi ? 8 : 0;
      const unsigned short* p = pw + cl * LDA + kb;
      fp.q[0] = *reinterpret_cast<const uint4*>(p);
      fp.q[1] = *reinterpret_cast<const uint4*>(p + 16);
    }
    // PV: B-fragment from pre-transposed V (rows = dv, contiguous keys)
#pragma unroll
    for (int c = 0; c < NC; ++c) {
      BF16Frag fv;
      int n  = dvOff + c * 16 + cl;
      int kb = hi ? 8 : 0;
      const unsigned short* p = Vz + (size_t)n * S + kt + kb;
      fv.q[0] = *reinterpret_cast<const uint4*>(p);
      fv.q[1] = *reinterpret_cast<const uint4*>(p + 16);
      acc[c] = __builtin_amdgcn_wmma_f32_16x16x32_bf16(false, fp.v, false, fv.v, (short)0, acc[c], false, false);
    }
  }

  // epilogue: normalize and scatter to O (heads interleaved via oLd/dvTotal)
#pragma unroll
  for (int c = 0; c < NC; ++c) {
    int dv = dvOff + c * 16 + cl;
#pragma unroll
    for (int r = 0; r < 8; ++r) {
      int i = q0 + r + hi * 8;
      O[(size_t)(b * S + i) * oLd + h * dvTotal + dv] = acc[c][r] / lrun[r];
    }
  }
}

// ---------------------------------------------------------------------------
// residual + LayerNorm: Y = LN(X + R) * g + b   (one block per row)
// ---------------------------------------------------------------------------
__global__ __launch_bounds__(256)
void add_ln_kernel(const float* __restrict__ X, const float* __restrict__ R,
                   const float* __restrict__ G, const float* __restrict__ Bb,
                   float* __restrict__ Y, int d) {
  __shared__ float sv[1024];
  __shared__ float red[256];
  const int row = blockIdx.x;
  const float* x = X + (size_t)row * d;
  const float* r = R + (size_t)row * d;
  float* y = Y + (size_t)row * d;

  float sum = 0.f;
  for (int c = threadIdx.x; c < d; c += 256) {
    float v = x[c] + r[c];
    sv[c] = v;
    sum += v;
  }
  red[threadIdx.x] = sum; __syncthreads();
  for (int s = 128; s > 0; s >>= 1) {
    if (threadIdx.x < s) red[threadIdx.x] += red[threadIdx.x + s];
    __syncthreads();
  }
  float mean = red[0] / (float)d;
  __syncthreads();

  float sq = 0.f;
  for (int c = threadIdx.x; c < d; c += 256) { float t = sv[c] - mean; sq += t * t; }
  red[threadIdx.x] = sq; __syncthreads();
  for (int s = 128; s > 0; s >>= 1) {
    if (threadIdx.x < s) red[threadIdx.x] += red[threadIdx.x + s];
    __syncthreads();
  }
  float inv = rsqrtf(red[0] / (float)d + 1e-5f);
  for (int c = threadIdx.x; c < d; c += 256)
    y[c] = (sv[c] - mean) * inv * G[c] + Bb[c];
}

// ---------------------------------------------------------------------------
// strided f32 -> bf16 convert (per b,h block):  dst[z][rows][cols]
// ---------------------------------------------------------------------------
__global__ void cvt_bf16_kernel(const float* __restrict__ src, int ld, int colOff,
                                int rows, int cols, unsigned short* __restrict__ dst,
                                int H) {
  const int z = blockIdx.z;
  const int b = z / H, h = z % H;
  const float* s = src + (size_t)b * rows * ld + colOff + (size_t)h * cols;
  unsigned short* d = dst + (size_t)z * rows * cols;
  const int n = rows * cols;
  for (int i = blockIdx.x * 256 + threadIdx.x; i < n; i += gridDim.x * 256) {
    int r = i / cols, c = i - r * cols;
    d[i] = f2bf(s[(size_t)r * ld + c]);
  }
}

// transpose + convert: dst[z][cols][rows]  (writes coalesced)
__global__ void cvt_bf16_t_kernel(const float* __restrict__ src, int ld, int colOff,
                                  int rows, int cols, unsigned short* __restrict__ dst,
                                  int H) {
  const int z = blockIdx.z;
  const int b = z / H, h = z % H;
  const float* s = src + (size_t)b * rows * ld + colOff + (size_t)h * cols;
  unsigned short* d = dst + (size_t)z * rows * cols;
  const int n = rows * cols;
  for (int i = blockIdx.x * 256 + threadIdx.x; i < n; i += gridDim.x * 256) {
    int c = i / rows, r = i - c * rows;   // dst index i = c*rows + r
    d[i] = f2bf(s[(size_t)r * ld + c]);
  }
}

// y = x + pos_emb[s mod S]
__global__ void add_pos_kernel(const float* __restrict__ x, const float* __restrict__ pos,
                               float* __restrict__ y, int S, int d, int total) {
  for (int i = blockIdx.x * 256 + threadIdx.x; i < total; i += gridDim.x * 256) {
    int col = i % d;
    int s = (i / d) % S;
    y[i] = x[i] + pos[(size_t)s * d + col];
  }
}

// ---------------------------------------------------------------------------
// Host orchestration
// ---------------------------------------------------------------------------
extern "C" void kernel_launch(void* const* d_in, const int* in_sizes, int n_in,
                              void* d_out, int out_size, void* d_ws, size_t ws_size,
                              hipStream_t stream) {
  (void)in_sizes; (void)n_in; (void)out_size; (void)ws_size;

  constexpr int Bn = 2, S = 2048, D0 = 1024;
  constexpr int M = Bn * S;                  // 4096 rows, multiple of 128
  static const int HEADS_[6] = {16, 8, 4, 1, 1, 1};

  auto fp = [&](int i) { return (const float*)d_in[i]; };
  const float* x = fp(0);
  // input flat order (setup_inputs insertion order):
  // 0: x; 1..288: levels[i][l]{qkv.W,qkv.b,out.W,out.b,ln1_g,ln1_b,ff1.W,ff1.b,ff2.W,ff2.b,ln2_g,ln2_b}
  // 289..298: compressors {W,b}x5 ; 299..308: decompressors {W,b}x5
  // 309: pos_emb; 310..317: q.W,q.b,k.W,k.b,v.W,v.b,o.W,o.b

  // workspace carve-up
  float* ACT0 = (float*)d_ws;                                   // 4096x1024 f32
  float* ACT1 = ACT0 + (size_t)M * D0;
  float* BIG0 = ACT1 + (size_t)M * D0;                          // up to 4096x4096 f32
  float* BIG1 = BIG0 + (size_t)M * 4096;                        // 4096x1024 f32
  unsigned short* QBF  = (unsigned short*)(BIG1 + (size_t)M * D0);
  unsigned short* KBF  = QBF + (size_t)M * D0;
  unsigned short* VTBF = KBF + (size_t)M * D0;

  auto gemm = [&](const float* A, const float* W, const float* bias, float* C,
                  int M_, int N_, int K_, int relu) {
    dim3 g((N_ + 63) / 64, M_ / 128, 1);
    gemm_bias_kernel<<<g, dim3(256, 1, 1), 0, stream>>>(A, W, bias, C, M_, N_, K_, relu);
  };

  float* actbuf[2] = {ACT0, ACT1};
  int pb = 0;
  const float* cur = x;
  int d = D0, ffdim = 4096;

  for (int lev = 0; lev < 6; ++lev) {
    const int H = HEADS_[lev];
    const int hd = d / H;
    const float qscale = 1.0f / sqrtf((float)hd);
    for (int ly = 0; ly < 4; ++ly) {
      const int pi = 1 + (lev * 4 + ly) * 12;
      const float *qkvW = fp(pi + 0), *qkvb = fp(pi + 1);
      const float *outW = fp(pi + 2), *outb = fp(pi + 3);
      const float *ln1g = fp(pi + 4), *ln1b = fp(pi + 5);
      const float *ff1W = fp(pi + 6), *ff1b = fp(pi + 7);
      const float *ff2W = fp(pi + 8), *ff2b = fp(pi + 9);
      const float *ln2g = fp(pi + 10), *ln2b = fp(pi + 11);

      // QKV projection
      gemm(cur, qkvW, qkvb, BIG0, M, 3 * d, d, 0);

      // pack per-head bf16 Q/K and transposed V
      const int Z = Bn * H;
      dim3 cg((S * hd + 255) / 256, 1, Z);
      cvt_bf16_kernel  <<<cg, 256, 0, stream>>>(BIG0, 3 * d, 0,     S, hd, QBF,  H);
      cvt_bf16_kernel  <<<cg, 256, 0, stream>>>(BIG0, 3 * d, d,     S, hd, KBF,  H);
      cvt_bf16_t_kernel<<<cg, 256, 0, stream>>>(BIG0, 3 * d, 2 * d, S, hd, VTBF, H);

      // flash MHA -> BIG1 [M, d] (heads interleaved)
      dim3 fg(S / 64, 1, Z);
      if (hd == 32)
        flash_kernel<2, false><<<fg, 128, 0, stream>>>(QBF, KBF, VTBF, BIG1, S, hd, hd, H, d, qscale, 1, 1);
      else if (hd == 64)
        flash_kernel<4, false><<<fg, 128, 0, stream>>>(QBF, KBF, VTBF, BIG1, S, hd, hd, H, d, qscale, 1, 1);
      else // 128
        flash_kernel<8, false><<<fg, 128, 0, stream>>>(QBF, KBF, VTBF, BIG1, S, hd, hd, H, d, qscale, 1, 1);

      // output projection, residual + LN1
      gemm(BIG1, outW, outb, BIG0, M, d, d, 0);
      add_ln_kernel<<<M, 256, 0, stream>>>(cur, BIG0, ln1g, ln1b, actbuf[pb], d);
      cur = actbuf[pb]; pb ^= 1;

      // FFN, residual + LN2
      gemm(cur, ff1W, ff1b, BIG0, M, ffdim, d, 1);
      gemm(BIG0, ff2W, ff2b, BIG1, M, d, ffdim, 0);
      add_ln_kernel<<<M, 256, 0, stream>>>(cur, BIG1, ln2g, ln2b, actbuf[pb], d);
      cur = actbuf[pb]; pb ^= 1;
    }
    if (lev < 5) {
      gemm(cur, fp(289 + 2 * lev), fp(290 + 2 * lev), actbuf[pb], M, d / 2, d, 0);
      cur = actbuf[pb]; pb ^= 1;
      d >>= 1; ffdim >>= 1;
    }
  }

  // decompress 32 -> 1024
  for (int i = 4; i >= 0; --i) {
    const int outd = 1024 >> i, ind = 1024 >> (i + 1);
    gemm(cur, fp(299 + 2 * i), fp(300 + 2 * i), actbuf[pb], M, outd, ind, 0);
    cur = actbuf[pb]; pb ^= 1;
  }

  // final sparse attention (d_model = 1024, single head)
  float* qIn = actbuf[pb]; pb ^= 1;
  add_pos_kernel<<<(M * D0) / 256, 256, 0, stream>>>(cur, fp(309), qIn, S, D0, M * D0);

  dim3 cg2(((S * D0) + 255) / 256, 1, Bn);
  gemm(qIn, fp(310), fp(311), BIG1, M, D0, D0, 0);
  cvt_bf16_kernel  <<<cg2, 256, 0, stream>>>(BIG1, D0, 0, S, D0, QBF, 1);
  gemm(x,   fp(312), fp(313), BIG0, M, D0, D0, 0);
  cvt_bf16_kernel  <<<cg2, 256, 0, stream>>>(BIG0, D0, 0, S, D0, KBF, 1);
  gemm(x,   fp(314), fp(315), BIG0, M, D0, D0, 0);
  cvt_bf16_t_kernel<<<cg2, 256, 0, stream>>>(BIG0, D0, 0, S, D0, VTBF, 1);

  // window 512 (half 256), stride 2048/16 = 128; scale = 1/sqrt(1024)
  dim3 fg2(S / 64, D0 / 128, Bn);
  flash_kernel<8, true><<<fg2, 128, 0, stream>>>(QBF, KBF, VTBF, BIG1,
                                                 S, D0, D0, 1, D0,
                                                 1.0f / 32.0f, 256, 128);

  // output projection -> d_out
  gemm(BIG1, fp(316), fp(317), (float*)d_out, M, D0, D0, 0);
}
